// DecoderSimpleAn_9689446220121
// MI455X (gfx1250) — matrix operand order repaired and checked by reading, not compiled, not run on hardware
//
#include <hip/hip_runtime.h>
#include <stdint.h>

// ---------------- problem constants ----------------
#define BB     256
#define SS     500
#define HH     512
#define DNN    256
#define DTT    512
#define VSTATc 1000
#define VDc    9000
#define M0c    4096
#define DEROWS 4352            // M0 + B
#define SROW   512             // padded score row stride
#define WTCOLS 10000           // VSTAT + VD

typedef __attribute__((ext_vector_type(16))) __bf16 v16bf;
typedef __attribute__((ext_vector_type(8)))  float  v8f;

// ---------------- fragment loaders (f32 memory -> bf16 WMMA fragments) ----------------
// Hardware f32->bf16 conversion (v_cvt_pk_bf16_f32) via native __bf16 casts.
// Edge handling: callers CLAMP the row/col index instead of zero-filling;
// out-of-range lanes only feed output rows/cols that the epilogue never stores,
// so in-bounds-but-garbage data is safe and keeps the K-loop branch-free.

// A fragment: M x K row-major source. Lane l -> row (l&15), K chunks
// {kb..kb+7} and {kb+16..kb+23} with kb = 8*(l>>4).   (ISA 7.12.2, 16-bit A 16x32)
__device__ __forceinline__ v16bf load_frag_A(const float* __restrict__ A, int lda,
                                             int row, int kk, int lane) {
  const float* p = A + (size_t)row * lda + kk + ((lane >> 4) << 3);
  union { v16bf v; __bf16 b[16]; } u;
#pragma unroll
  for (int i = 0; i < 8; ++i) u.b[i]     = (__bf16)p[i];
#pragma unroll
  for (int i = 0; i < 8; ++i) u.b[8 + i] = (__bf16)p[16 + i];
  return u.v;
}

// B fragment: B stored N x K row-major (so B^T is K x N). Lane l -> column
// (l&15), K chunks {kb..kb+7} and {kb+16..kb+23} with kb = 8*(l>>4) (same as A).
__device__ __forceinline__ v16bf load_frag_B(const float* __restrict__ Bm, int ldb,
                                             int col, int kk, int lane) {
  const float* p = Bm + (size_t)col * ldb + kk + ((lane >> 4) << 3);
  union { v16bf v; __bf16 b[16]; } u;
#pragma unroll
  for (int i = 0; i < 8; ++i) u.b[i]     = (__bf16)p[i];
#pragma unroll
  for (int i = 0; i < 8; ++i) u.b[8 + i] = (__bf16)p[16 + i];
  return u.v;
}

__device__ __forceinline__ float sigf(float x) { return 1.f / (1.f + __expf(-x)); }

// ---------------- generic WMMA GEMM:  C = [C +] A(MxK) * B(NxK)^T + bias ----------------
template<int ACT>   // 0 = none, 1 = tanh
__global__ void gemm_bf16(const float* __restrict__ A, const float* __restrict__ Bm,
                          const float* __restrict__ bias, float* __restrict__ C,
                          int M, int N, int K, int ldc, int accumulate) {
  const int lane = threadIdx.x & 31;
  const int wid  = (blockIdx.x * blockDim.x + threadIdx.x) >> 5;
  const int nCh  = (N + 63) >> 6;
  const int tot  = (M >> 4) * nCh;
  if (wid >= tot) return;
  const int m0 = (wid / nCh) << 4;
  const int n0 = (wid % nCh) << 6;

  // clamped (in-bounds) column per accumulator; garbage columns are never stored
  int ncol[4];
#pragma unroll
  for (int t = 0; t < 4; ++t) {
    int n = n0 + 16 * t + (lane & 15);
    ncol[t] = (n < N) ? n : (N - 1);
  }

  v8f acc[4] = {};
  const int arow = m0 + (lane & 15);
  for (int kk = 0; kk < K; kk += 32) {
    v16bf a = load_frag_A(A, K, arow, kk, lane);
#pragma unroll
    for (int t = 0; t < 4; ++t) {
      v16bf b = load_frag_B(Bm, K, ncol[t], kk, lane);
      acc[t] = __builtin_amdgcn_wmma_f32_16x16x32_bf16(false, a, false, b,
                                                       (short)0, acc[t], false, false);
    }
  }
  const int mbase = m0 + ((lane >> 4) << 3);   // D element (M = j + 8*(l>>4), N = l&15)
#pragma unroll
  for (int t = 0; t < 4; ++t) {
    int n = n0 + 16 * t + (lane & 15);
    if (n >= N) continue;
    float bv = bias ? bias[n] : 0.f;
#pragma unroll
    for (int j = 0; j < 8; ++j) {
      size_t idx = (size_t)(mbase + j) * ldc + n;
      float v = acc[t][j] + bv;
      if (accumulate) v += C[idx];
      if (ACT == 1) v = tanhf(v);
      C[idx] = v;
    }
  }
}

// ---------------- fused attention scores ----------------
// grid = B * ceil(S/16) blocks of 256 threads (8 waves; wave w covers N columns w*64..)
// scores[b,s] = v . tanh(sc[b,:] + Wm_b + enc[b,s,:] @ Wm^T) + v_b  (masked)
__global__ void attn_scores(const float* __restrict__ enc, const float* __restrict__ Wm,
                            const float* __restrict__ Wm_b, const float* __restrict__ sc,
                            const float* __restrict__ v_w, const float* __restrict__ v_b,
                            const uint8_t* __restrict__ mask, float* __restrict__ scores) {
  const int b    = blockIdx.x >> 5;
  const int s0   = (blockIdx.x & 31) << 4;
  const int lane = threadIdx.x & 31;
  const int n0   = (threadIdx.x >> 5) << 6;

  __shared__ float ssum[16];
  if (threadIdx.x < 16) ssum[threadIdx.x] = 0.f;
  __syncthreads();

  const float* E = enc + (size_t)b * SS * HH;
  int row = s0 + (lane & 15);
  if (row >= SS) row = SS - 1;      // clamp: rows >= S feed score slots never stored

  v8f acc[4] = {};
  for (int kk = 0; kk < HH; kk += 32) {
    v16bf a = load_frag_A(E, HH, row, kk, lane);
#pragma unroll
    for (int t = 0; t < 4; ++t) {
      v16bf bb = load_frag_B(Wm, HH, n0 + 16 * t + (lane & 15), kk, lane);
      acc[t] = __builtin_amdgcn_wmma_f32_16x16x32_bf16(false, a, false, bb,
                                                       (short)0, acc[t], false, false);
    }
  }
  float pj[8];
#pragma unroll
  for (int j = 0; j < 8; ++j) pj[j] = 0.f;
#pragma unroll
  for (int t = 0; t < 4; ++t) {
    int n = n0 + 16 * t + (lane & 15);
    float vn  = v_w[n];
    float scb = sc[(size_t)b * HH + n] + Wm_b[n];
#pragma unroll
    for (int j = 0; j < 8; ++j) pj[j] += vn * tanhf(scb + acc[t][j]);
  }
  const int mbase = (lane >> 4) << 3;
#pragma unroll
  for (int j = 0; j < 8; ++j) atomicAdd(&ssum[mbase + j], pj[j]);
  __syncthreads();
  if (threadIdx.x < 16) {
    int s = s0 + threadIdx.x;
    if (s < SS) {
      float val = ssum[threadIdx.x] + v_b[0];
      if (mask[(size_t)b * SS + s]) val = -1e20f;
      scores[(size_t)b * SROW + s] = val;
    }
  }
}

// ---------------- control: per-batch routing + scan for new slots ----------------
__global__ void control_k(const int* __restrict__ t_input, const long long* __restrict__ lt,
                          int* dyn_dense, int* safe_idx, int* widx, int* slot_new,
                          int* is_new_a, int* is_update_a, int* is_static_a) {
  int b = threadIdx.x;
  int t = t_input[b];
  int is_static = (t < VSTATc);
  int dd = is_static ? 0 : (t - VSTATc);
  long long lok = is_static ? -1ll : lt[(size_t)b * VDc + dd];
  int is_update = (!is_static) && (lok != -1ll);
  int is_new    = (!is_static) && (lok == -1ll);
  int sidx = is_update ? (int)lok : 0;

  __shared__ int scan[BB];
  scan[b] = is_new;
  __syncthreads();
  for (int o = 1; o < BB; o <<= 1) {          // inclusive Hillis-Steele scan
    int v = (b >= o) ? scan[b - o] : 0;
    __syncthreads();
    scan[b] += v;
    __syncthreads();
  }
  int sn = M0c + scan[b] - 1;
  dyn_dense[b]   = dd;
  safe_idx[b]    = sidx;
  is_new_a[b]    = is_new;
  is_update_a[b] = is_update;
  is_static_a[b] = is_static;
  slot_new[b]    = sn;
  widx[b] = is_update ? sidx : (is_new ? sn : DEROWS);   // DEROWS == dummy (dropped)
}

// ---------------- small data-movement kernels ----------------
__global__ void build_x1(const int* __restrict__ n_input, const int* __restrict__ t_input,
                         const int* __restrict__ is_static_a, const int* __restrict__ is_update_a,
                         const int* __restrict__ dyn_dense, const int* __restrict__ safe_idx,
                         const float* __restrict__ embN, const float* __restrict__ embT,
                         const float* __restrict__ de_h, const float* __restrict__ dyn_h,
                         float* __restrict__ x1) {
  int b = blockIdx.x;
  int ni = n_input[b];
  for (int j = threadIdx.x; j < DNN; j += blockDim.x)
    x1[(size_t)b * 768 + j] = embN[(size_t)ni * DNN + j];
  const float* src;
  if (is_static_a[b])      src = embT  + (size_t)t_input[b]  * DTT;
  else if (is_update_a[b]) src = de_h  + (size_t)safe_idx[b] * DTT;
  else                     src = dyn_h + (size_t)dyn_dense[b] * DTT;
  for (int j = threadIdx.x; j < DTT; j += blockDim.x)
    x1[(size_t)b * 768 + DNN + j] = src[j];
}

__global__ void build_xd(const float* __restrict__ x1, const float* __restrict__ h0,
                         float* __restrict__ xd) {
  int b = blockIdx.x, j = threadIdx.x;               // 512 threads
  if (j < DNN) xd[(size_t)b * 768 + j] = x1[(size_t)b * 768 + j];
  xd[(size_t)b * 768 + DNN + j] = h0[(size_t)b * HH + j];
}

__global__ void build_hd0cd0(const int* __restrict__ is_update_a, const int* __restrict__ is_new_a,
                             const int* __restrict__ safe_idx, const int* __restrict__ dyn_dense,
                             const float* __restrict__ de_h, const float* __restrict__ de_c,
                             const float* __restrict__ dyn_h, const float* __restrict__ dyn_c,
                             float* __restrict__ hd0, float* __restrict__ cd0) {
  int b = blockIdx.x, j = threadIdx.x;               // 512 threads
  float hv = 0.f, cv = 0.f;
  if (is_update_a[b]) {
    hv = de_h[(size_t)safe_idx[b] * DTT + j];
    cv = de_c[(size_t)safe_idx[b] * DTT + j];
  } else if (is_new_a[b]) {
    hv = dyn_h[(size_t)dyn_dense[b] * DTT + j];
    cv = dyn_c[(size_t)dyn_dense[b] * DTT + j];
  }
  hd0[(size_t)b * DTT + j] = hv;
  cd0[(size_t)b * DTT + j] = cv;
}

__global__ void lstm_elem(const float* __restrict__ g, const float* __restrict__ c0,
                          float* __restrict__ hO, float* __restrict__ cO) {
  int i = blockIdx.x * blockDim.x + threadIdx.x;
  if (i >= BB * HH) return;
  int b = i >> 9, hh = i & 511;
  const float* gb = g + (size_t)b * 2048;
  float ig = gb[hh], fg = gb[512 + hh], gg = gb[1024 + hh], og = gb[1536 + hh];
  float c2 = sigf(fg) * c0[i] + sigf(ig) * tanhf(gg);
  cO[i] = c2;
  hO[i] = sigf(og) * tanhf(c2);
}

__global__ void softmax_k(float* __restrict__ scores) {
  int b = blockIdx.x, t = threadIdx.x;               // 512 threads
  __shared__ float red[512];
  float x = (t < SS) ? scores[(size_t)b * SROW + t] : -INFINITY;
  red[t] = x; __syncthreads();
  for (int o = 256; o; o >>= 1) { if (t < o) red[t] = fmaxf(red[t], red[t + o]); __syncthreads(); }
  float mx = red[0]; __syncthreads();
  float e = (t < SS) ? __expf(x - mx) : 0.f;
  red[t] = e; __syncthreads();
  for (int o = 256; o; o >>= 1) { if (t < o) red[t] += red[t + o]; __syncthreads(); }
  float inv = 1.f / red[0];
  if (t < SS) scores[(size_t)b * SROW + t] = e * inv;
}

__global__ void context_k(const float* __restrict__ aw, const float* __restrict__ enc,
                          float* __restrict__ ctx) {
  int b = blockIdx.x, h = threadIdx.x;               // 512 threads
  const float* E = enc + (size_t)b * SS * HH + h;
  float s = 0.f;
  for (int i = 0; i < SS; ++i) s += aw[(size_t)b * SROW + i] * E[(size_t)i * HH];
  ctx[(size_t)b * HH + h] = s;
}

__global__ void build_cat3(const float* __restrict__ ctx, const float* __restrict__ h1,
                           const float* __restrict__ hp, float* __restrict__ cat3) {
  int b = blockIdx.x, j = threadIdx.x;               // 512 threads
  cat3[(size_t)b * 1536 + j]        = ctx[(size_t)b * HH + j];
  cat3[(size_t)b * 1536 + 512 + j]  = h1[(size_t)b * HH + j];
  cat3[(size_t)b * 1536 + 1024 + j] = hp[(size_t)b * HH + j];
}

__global__ void de_init(const float* __restrict__ src, float* __restrict__ dst) {
  size_t i = (size_t)blockIdx.x * blockDim.x + threadIdx.x;
  if (i >= (size_t)DEROWS * DTT) return;
  dst[i] = (i < (size_t)M0c * DTT) ? src[i] : 0.f;
}

__global__ void scatter_de(const int* __restrict__ widx, const float* __restrict__ hd,
                           const float* __restrict__ cd, float* __restrict__ deh,
                           float* __restrict__ dec) {
  int b = blockIdx.x, j = threadIdx.x;               // 512 threads
  int w = widx[b];
  if (w < DEROWS) {
    deh[(size_t)w * DTT + j] = hd[(size_t)b * DTT + j];
    dec[(size_t)w * DTT + j] = cd[(size_t)b * DTT + j];
  }
}

__global__ void lt_init(const long long* __restrict__ ltin, float* __restrict__ ltf) {
  size_t i = (size_t)blockIdx.x * blockDim.x + threadIdx.x;
  if (i >= (size_t)BB * VDc) return;
  ltf[i] = (float)ltin[i];
}

__global__ void lt_scatter(const int* __restrict__ is_new_a, const int* __restrict__ dyn_dense,
                           const int* __restrict__ slot_new, float* __restrict__ ltf) {
  int b = threadIdx.x;
  if (is_new_a[b]) ltf[(size_t)b * VDc + dyn_dense[b]] = (float)slot_new[b];
}

__global__ void wdyn_gather(const float* __restrict__ ltf, const float* __restrict__ sbuf,
                            float* __restrict__ wt) {
  size_t i = (size_t)blockIdx.x * blockDim.x + threadIdx.x;
  if (i >= (size_t)BB * VDc) return;
  int b = (int)(i / VDc), j = (int)(i % VDc);
  float lv = ltf[i];
  if (lv >= 0.f) {
    int v = (int)lv;
    wt[(size_t)b * WTCOLS + VSTATc + j] = sbuf[(size_t)b * DEROWS + v];
  }
}

// ---------------- host orchestration ----------------
extern "C" void kernel_launch(void* const* d_in, const int* in_sizes, int n_in,
                              void* d_out, int out_size, void* d_ws, size_t ws_size,
                              hipStream_t stream) {
  (void)in_sizes; (void)n_in; (void)out_size; (void)ws_size;

  const int*       n_input  = (const int*)d_in[0];
  const int*       t_input  = (const int*)d_in[1];
  const float*     h0       = (const float*)d_in[2];
  const float*     c0       = (const float*)d_in[3];
  const float*     enc      = (const float*)d_in[4];
  const uint8_t*   maskp    = (const uint8_t*)d_in[5];
  const float*     h_parent = (const float*)d_in[6];
  const long long* lookup   = (const long long*)d_in[7];
  const float*     de_h     = (const float*)d_in[8];
  const float*     de_c     = (const float*)d_in[9];
  const float*     embN     = (const float*)d_in[10];
  const float*     embT     = (const float*)d_in[11];
  const float*     W_ih     = (const float*)d_in[12];
  const float*     W_hh     = (const float*)d_in[13];
  const float*     b_ih     = (const float*)d_in[14];
  const float*     b_hh     = (const float*)d_in[15];
  const float*     W_ih_d   = (const float*)d_in[16];
  const float*     W_hh_d   = (const float*)d_in[17];
  const float*     b_ih_d   = (const float*)d_in[18];
  const float*     b_hh_d   = (const float*)d_in[19];
  const float*     Wo_w     = (const float*)d_in[20];
  const float*     Wo_b     = (const float*)d_in[21];
  const float*     Wh_w     = (const float*)d_in[22];
  const float*     Wh_b     = (const float*)d_in[23];
  const float*     Wm_w     = (const float*)d_in[24];
  const float*     Wm_b     = (const float*)d_in[25];
  const float*     v_w      = (const float*)d_in[26];
  const float*     v_b      = (const float*)d_in[27];
  const float*     dyn_h    = (const float*)d_in[28];
  const float*     dyn_c    = (const float*)d_in[29];

  // ---- output layout (floats, return order) ----
  float* out_f = (float*)d_out;
  float* wt      = out_f;                                     // [B, 10000]
  float* outv    = wt   + (size_t)BB * WTCOLS;                // [B, 512]
  float* h1      = outv + (size_t)BB * HH;                    // [B, 512]
  float* c1      = h1   + (size_t)BB * HH;                    // [B, 512]
  float* ltf     = c1   + (size_t)BB * HH;                    // [B, 9000] (as float)
  float* de_h_n  = ltf  + (size_t)BB * VDc;                   // [4352, 512]
  float* de_c_n  = de_h_n + (size_t)DEROWS * DTT;             // [4352, 512]

  // ---- workspace layout ----
  int* wi = (int*)d_ws;
  int* dyn_dense = wi;          int* safe_idx  = wi + 256;
  int* widx      = wi + 512;    int* slot_new  = wi + 768;
  int* is_new_a  = wi + 1024;   int* is_upd_a  = wi + 1280;
  int* is_stat_a = wi + 1536;
  float* fw = (float*)((char*)d_ws + 8192);
  float* x1    = fw;                         // 256*768
  float* g1    = x1   + 196608;              // 256*2048
  float* sc    = g1   + 524288;              // 256*512
  float* sco   = sc   + 131072;              // 256*512 (scores -> aw in place)
  float* ctx   = sco  + 131072;              // 256*512
  float* cat3  = ctx  + 131072;              // 256*1536
  float* xd    = cat3 + 393216;              // 256*768
  float* hd0   = xd   + 196608;              // 256*512
  float* cd0   = hd0  + 131072;              // 256*512
  float* gd    = cd0  + 131072;              // 256*2048
  float* hd    = gd   + 524288;              // 256*512
  float* cd    = hd   + 131072;              // 256*512
  float* sbuf  = cd   + 131072;              // 256*4352

  auto gemm = [&](const float* A, const float* Bm, const float* bias, float* C,
                  int M, int N, int K, int ldc, int acc, int act) {
    int nCh = (N + 63) / 64;
    int tot = (M / 16) * nCh;
    int blocks = (tot + 7) / 8;
    if (act) hipLaunchKernelGGL((gemm_bf16<1>), dim3(blocks), dim3(256), 0, stream,
                                A, Bm, bias, C, M, N, K, ldc, acc);
    else     hipLaunchKernelGGL((gemm_bf16<0>), dim3(blocks), dim3(256), 0, stream,
                                A, Bm, bias, C, M, N, K, ldc, acc);
  };

  // 1) routing / scan
  hipLaunchKernelGGL(control_k, dim3(1), dim3(256), 0, stream,
                     t_input, lookup, dyn_dense, safe_idx, widx, slot_new,
                     is_new_a, is_upd_a, is_stat_a);
  // 2) input concat for LSTM1
  hipLaunchKernelGGL(build_x1, dim3(BB), dim3(256), 0, stream,
                     n_input, t_input, is_stat_a, is_upd_a, dyn_dense, safe_idx,
                     embN, embT, de_h, dyn_h, x1);
  // 3) LSTM1: g1 = x1@W_ih^T + b_ih ; g1 += h0@W_hh^T + b_hh ; pointwise -> h1,c1
  gemm(x1, W_ih, b_ih, g1, BB, 2048, 768, 2048, 0, 0);
  gemm(h0, W_hh, b_hh, g1, BB, 2048, 512, 2048, 1, 0);
  hipLaunchKernelGGL(lstm_elem, dim3((BB * HH + 255) / 256), dim3(256), 0, stream,
                     g1, c0, h1, c1);
  // 4) attention
  gemm(h1, Wh_w, Wh_b, sc, BB, 512, 512, 512, 0, 0);
  hipLaunchKernelGGL(attn_scores, dim3(BB * 32), dim3(256), 0, stream,
                     enc, Wm_w, Wm_b, sc, v_w, v_b, maskp, sco);
  hipLaunchKernelGGL(softmax_k, dim3(BB), dim3(512), 0, stream, sco);
  hipLaunchKernelGGL(context_k, dim3(BB), dim3(512), 0, stream, sco, enc, ctx);
  // 5) output projection: out = tanh([ctx,h1,h_parent]@Wo^T + Wo_b)
  hipLaunchKernelGGL(build_cat3, dim3(BB), dim3(512), 0, stream, ctx, h1, h_parent, cat3);
  gemm(cat3, Wo_w, Wo_b, outv, BB, 512, 1536, 512, 0, 1);
  // 6) dynamic-embedding LSTM
  hipLaunchKernelGGL(build_xd, dim3(BB), dim3(512), 0, stream, x1, h0, xd);
  hipLaunchKernelGGL(build_hd0cd0, dim3(BB), dim3(512), 0, stream,
                     is_upd_a, is_new_a, safe_idx, dyn_dense, de_h, de_c, dyn_h, dyn_c,
                     hd0, cd0);
  gemm(xd, W_ih_d, b_ih_d, gd, BB, 2048, 768, 2048, 0, 0);
  gemm(hd0, W_hh_d, b_hh_d, gd, BB, 2048, 512, 2048, 1, 0);
  hipLaunchKernelGGL(lstm_elem, dim3((BB * HH + 255) / 256), dim3(256), 0, stream,
                     gd, cd0, hd, cd);
  // 7) state-table update
  int deBlocks = (int)(((size_t)DEROWS * DTT + 255) / 256);
  hipLaunchKernelGGL(de_init, dim3(deBlocks), dim3(256), 0, stream, de_h, de_h_n);
  hipLaunchKernelGGL(de_init, dim3(deBlocks), dim3(256), 0, stream, de_c, de_c_n);
  hipLaunchKernelGGL(scatter_de, dim3(BB), dim3(512), 0, stream, widx, hd, cd, de_h_n, de_c_n);
  // 8) lookup-table update (emitted as float)
  int ltBlocks = (int)(((size_t)BB * VDc + 255) / 256);
  hipLaunchKernelGGL(lt_init, dim3(ltBlocks), dim3(256), 0, stream, lookup, ltf);
  hipLaunchKernelGGL(lt_scatter, dim3(1), dim3(256), 0, stream,
                     is_new_a, dyn_dense, slot_new, ltf);
  // 9) logits: static, dynamic-init, then overwrite valid slots with table scores
  gemm(outv, embT,  nullptr, wt,           BB, VSTATc, 512, WTCOLS, 0, 0);
  gemm(outv, dyn_h, nullptr, wt + VSTATc,  BB, VDc,    512, WTCOLS, 0, 0);
  gemm(outv, de_h_n, nullptr, sbuf,        BB, DEROWS, 512, DEROWS, 0, 0);
  hipLaunchKernelGGL(wdyn_gather, dim3(ltBlocks), dim3(256), 0, stream, ltf, sbuf, wt);
}
